// Set2Set_65876208386530
// MI455X (gfx1250) — matrix operand order repaired and checked by reading, not compiled, not run on hardware
//
#include <hip/hip_runtime.h>
#include <hip/hip_bf16.h>
#include <math.h>

// ---- CDNA5 wave32 WMMA types ------------------------------------------------
typedef __bf16 bf16;
typedef __attribute__((ext_vector_type(8)))  bf16  bf16x8;
typedef __attribute__((ext_vector_type(16))) bf16  v16bf;
typedef __attribute__((ext_vector_type(8)))  float v8f;

// ---- utility ----------------------------------------------------------------
__global__ void s2s_zero_f32(float* p, long n) {
  long i = (long)blockIdx.x * blockDim.x + threadIdx.x;
  long s = (long)gridDim.x * blockDim.x;
  for (; i < n; i += s) p[i] = 0.f;
}

// segment starts: idx is sorted; start[g] = first node of graph g, start[G] = N
__global__ void s2s_seg_start(const long long* __restrict__ idx, int* __restrict__ start,
                              int N, int G) {
  int n = blockIdx.x * blockDim.x + threadIdx.x;
  if (n >= N) return;
  int cur  = (int)idx[n];
  int prev = (n == 0) ? -1 : (int)idx[n - 1];
  for (int g = prev + 1; g <= cur; ++g) start[g] = n;
  if (n == N - 1)
    for (int g = cur + 1; g <= G; ++g) start[g] = N;
}

// W^T [N4 x K3] split into bf16 hi+lo (rows = output col n, contiguous k)
__global__ void s2s_cvt_weights(const float* __restrict__ kern, const float* __restrict__ rkern,
                                bf16* __restrict__ Whi, bf16* __restrict__ Wlo,
                                int F, int K3, int N4) {
  long t = (long)blockIdx.x * blockDim.x + threadIdx.x;
  long total = (long)N4 * K3;
  if (t >= total) return;
  int n = (int)(t / K3);
  int k = (int)(t % K3);
  float w = (k < 2 * F) ? kern[(long)k * N4 + n] : rkern[(long)(k - 2 * F) * N4 + n];
  bf16 hi = (bf16)w;
  bf16 lo = (bf16)(w - (float)hi);
  Whi[t] = hi;
  Wlo[t] = lo;
}

// A = [q_star | h] as [Gp x K3] bf16 hi+lo, zero-padded rows for g >= G
__global__ void s2s_cvt_A(const float* __restrict__ qstar, const float* __restrict__ h,
                          bf16* __restrict__ Ahi, bf16* __restrict__ Alo,
                          int F, int K3, int G, int Gp) {
  long t = (long)blockIdx.x * blockDim.x + threadIdx.x;
  long total = (long)Gp * K3;
  if (t >= total) return;
  int g = (int)(t / K3);
  int k = (int)(t % K3);
  float v = 0.f;
  if (g < G) v = (k < 2 * F) ? qstar[(long)g * 2 * F + k] : h[(long)g * F + (k - 2 * F)];
  bf16 hi = (bf16)v;
  bf16 lo = (bf16)(v - (float)hi);
  Ahi[t] = hi;
  Alo[t] = lo;
}

// A-fragment: lane holds row m=lane&15; k = kh..kh+15 (kh = 16*(lane>=16))
__device__ __forceinline__ v16bf s2s_load_afrag(const bf16* p) {
  bf16x8 lo = *(const bf16x8*)p;        // k0+kh .. +7
  bf16x8 hi = *(const bf16x8*)(p + 8);  // k0+kh+8 .. +15
  return __builtin_shufflevector(lo, hi, 0, 1, 2, 3, 4, 5, 6, 7, 8, 9, 10, 11, 12, 13, 14, 15);
}

// GEMM: Z[Gp x N4] = A[Gp x K3] * W^T[N4 x K3]^T, split-bf16 3-WMMA accumulation.
// Block = 4 waves; wave tile = 32M x 64N (2x4 tiles of 16x16). No divergence: EXEC all ones.
__global__ __launch_bounds__(128) void s2s_lstm_gemm(
    const bf16* __restrict__ Ahi, const bf16* __restrict__ Alo,
    const bf16* __restrict__ Whi, const bf16* __restrict__ Wlo,
    float* __restrict__ Z, int K3, int N4) {
  const int lane  = threadIdx.x & 31;
  const int wave  = threadIdx.x >> 5;
  const int mbase = blockIdx.y * 32;
  const int nbase = blockIdx.x * 256 + wave * 64;
  const int lm    = lane & 15;
  const int kha   = (lane & 16);      // 0 or 16 (A contiguous layout)
  const int khb   = (lane & 16);      // 0 or 16 (B contiguous layout)

  const v8f vzero = {0.f, 0.f, 0.f, 0.f, 0.f, 0.f, 0.f, 0.f};
  v8f acc[2][4];
#pragma unroll
  for (int mi = 0; mi < 2; ++mi)
#pragma unroll
    for (int ni = 0; ni < 4; ++ni) acc[mi][ni] = vzero;

  const bf16* arhi[2];
  const bf16* arlo[2];
#pragma unroll
  for (int mi = 0; mi < 2; ++mi) {
    long r = (long)(mbase + mi * 16 + lm) * K3;
    arhi[mi] = Ahi + r;
    arlo[mi] = Alo + r;
  }
  const bf16* brhi[4];
  const bf16* brlo[4];
#pragma unroll
  for (int ni = 0; ni < 4; ++ni) {
    long r = (long)(nbase + ni * 16 + lm) * K3;
    brhi[ni] = Whi + r;
    brlo[ni] = Wlo + r;
  }

  for (int k0 = 0; k0 < K3; k0 += 32) {
    v16bf ah[2], al[2], bh[4], bl[4];
#pragma unroll
    for (int mi = 0; mi < 2; ++mi) {
      ah[mi] = s2s_load_afrag(arhi[mi] + k0 + kha);
      al[mi] = s2s_load_afrag(arlo[mi] + k0 + kha);
    }
#pragma unroll
    for (int ni = 0; ni < 4; ++ni) {
      bh[ni] = *(const v16bf*)(brhi[ni] + k0 + khb);
      bl[ni] = *(const v16bf*)(brlo[ni] + k0 + khb);
    }
#pragma unroll
    for (int mi = 0; mi < 2; ++mi)
#pragma unroll
      for (int ni = 0; ni < 4; ++ni) {
        acc[mi][ni] = __builtin_amdgcn_wmma_f32_16x16x32_bf16(
            false, ah[mi], false, bh[ni], (short)0, acc[mi][ni], false, false);
        acc[mi][ni] = __builtin_amdgcn_wmma_f32_16x16x32_bf16(
            false, ah[mi], false, bl[ni], (short)0, acc[mi][ni], false, false);
        acc[mi][ni] = __builtin_amdgcn_wmma_f32_16x16x32_bf16(
            false, al[mi], false, bh[ni], (short)0, acc[mi][ni], false, false);
      }
  }

  // C/D layout: VGPR r -> row = r + 8*(lane>=16); col = lane&15
  const int rofs = (lane >> 4) << 3;
#pragma unroll
  for (int mi = 0; mi < 2; ++mi)
#pragma unroll
    for (int ni = 0; ni < 4; ++ni) {
      long base = (long)(mbase + mi * 16 + rofs) * N4 + (nbase + ni * 16 + lm);
#pragma unroll
      for (int r = 0; r < 8; ++r) Z[base + (long)r * N4] = acc[mi][ni][r];
    }
}

// LSTM gates (Keras order i,f,g,o); writes h,c and q into q_star[:, :F]
__global__ void s2s_lstm_gates(const float* __restrict__ Z, const float* __restrict__ bias,
                               float* __restrict__ h, float* __restrict__ c,
                               float* __restrict__ qstar, int F, int N4, int G) {
  long t = (long)blockIdx.x * blockDim.x + threadIdx.x;
  if (t >= (long)G * F) return;
  int g = (int)(t / F);
  int f = (int)(t % F);
  const float* zr = Z + (long)g * N4;
  float zi = zr[f]         + bias[f];
  float zf = zr[F + f]     + bias[F + f];
  float zg = zr[2 * F + f] + bias[2 * F + f];
  float zo = zr[3 * F + f] + bias[3 * F + f];
  float ig = 1.f / (1.f + expf(-zi));
  float fg = 1.f / (1.f + expf(-zf));
  float gg = tanhf(zg);
  float og = 1.f / (1.f + expf(-zo));
  float cn = fg * c[t] + ig * gg;
  float hn = og * tanhf(cn);
  c[t] = cn;
  h[t] = hn;
  qstar[(long)g * 2 * F + f] = hn;
}

// e[n] = <x_n, q_{idx[n]}>; one wave32 per node, coalesced over features
__global__ void s2s_attn_dot(const float* __restrict__ x, const long long* __restrict__ idx,
                             const float* __restrict__ h, float* __restrict__ e,
                             int N, int F) {
  int lane = threadIdx.x & 31;
  int wib  = threadIdx.x >> 5;
  int node = blockIdx.x * (blockDim.x >> 5) + wib;
  if (node >= N) return;
  int g = (int)idx[node];
  const float* xr = x + (long)node * F;
  const float* qr = h + (long)g * F;
  float acc = 0.f;
  for (int j = lane; j < F; j += 32) acc += xr[j] * qr[j];
#pragma unroll
  for (int off = 16; off; off >>= 1) acc += __shfl_xor(acc, off, 32);
  if (lane == 0) e[node] = acc;
}

// per-graph segment softmax stats: emax (finite-guarded) and 1/denom (denom==0 -> 1)
__global__ void s2s_seg_stats(const float* __restrict__ e, const int* __restrict__ start,
                              float* __restrict__ emax, float* __restrict__ invden) {
  __shared__ float red[128];
  int g = blockIdx.x, tid = threadIdx.x;
  int s = start[g], t = start[g + 1];
  float m = -INFINITY;
  for (int n = s + tid; n < t; n += 128) m = fmaxf(m, e[n]);
  red[tid] = m;
  __syncthreads();
  for (int off = 64; off; off >>= 1) {
    if (tid < off) red[tid] = fmaxf(red[tid], red[tid + off]);
    __syncthreads();
  }
  float M = (t > s) ? red[0] : 0.f;
  __syncthreads();
  float sm = 0.f;
  for (int n = s + tid; n < t; n += 128) sm += expf(e[n] - M);
  red[tid] = sm;
  __syncthreads();
  for (int off = 64; off; off >>= 1) {
    if (tid < off) red[tid] += red[tid + off];
    __syncthreads();
  }
  if (tid == 0) {
    emax[g] = M;
    float d = red[0];
    invden[g] = (d == 0.f) ? 1.f : (1.f / d);
  }
}

// r[g] = sum_n a_n * x_n; block per graph, a staged in LDS in chunks of 128 nodes
__global__ void s2s_readout(const float* __restrict__ x, const float* __restrict__ e,
                            const int* __restrict__ start, const float* __restrict__ emax,
                            const float* __restrict__ invden, float* __restrict__ out,
                            int F) {
  __shared__ float a_sh[128];
  int g = blockIdx.x, tid = threadIdx.x; // blockDim = 256
  int s = start[g], t = start[g + 1];
  float M = emax[g], inv = invden[g];
  for (int fbase = 0; fbase < F; fbase += 256) {
    int f = fbase + tid;
    float acc = 0.f;
    for (int cs = s; cs < t; cs += 128) {
      int cnt = min(128, t - cs);
      __syncthreads();
      if (tid < cnt) a_sh[tid] = expf(e[cs + tid] - M) * inv;
      __syncthreads();
      if (f < F) {
        const float* xp = x + (long)cs * F + f;
        for (int j = 0; j < cnt; ++j) {
          __builtin_prefetch(xp + (long)(j + 8) * F, 0, 1);
          acc += a_sh[j] * xp[(long)j * F];
        }
      }
    }
    if (f < F) out[(long)g * 2 * F + F + f] = acc;
  }
}

// ---- host driver ------------------------------------------------------------
extern "C" void kernel_launch(void* const* d_in, const int* in_sizes, int n_in,
                              void* d_out, int out_size, void* d_ws, size_t ws_size,
                              hipStream_t stream) {
  const float*     x     = (const float*)d_in[0];
  const long long* idx   = (const long long*)d_in[1];
  const float*     kern  = (const float*)d_in[2];
  const float*     rkern = (const float*)d_in[3];
  const float*     bias  = (const float*)d_in[4];

  const int N  = in_sizes[1];
  const int F  = in_sizes[0] / N;          // 512
  const int G  = out_size / (2 * F);       // 1000
  const int K3 = 3 * F;                    // 1536
  const int N4 = 4 * F;                    // 2048
  const int Gp = ((G + 31) / 32) * 32;     // 1024

  float* qstar = (float*)d_out;

  char* ws = (char*)d_ws;
  size_t off = 0;
  auto carve = [&](size_t bytes) -> char* {
    char* p = ws + off;
    off += (bytes + 255) & ~(size_t)255;
    return p;
  };
  float* h      = (float*)carve((size_t)G * F * 4);
  float* c      = (float*)carve((size_t)G * F * 4);
  float* e      = (float*)carve((size_t)N * 4);
  int*   start  = (int*)carve((size_t)(G + 1) * 4);
  float* emax   = (float*)carve((size_t)G * 4);
  float* invden = (float*)carve((size_t)G * 4);
  bf16*  Ahi    = (bf16*)carve((size_t)Gp * K3 * 2);
  bf16*  Alo    = (bf16*)carve((size_t)Gp * K3 * 2);
  bf16*  Whi    = (bf16*)carve((size_t)N4 * K3 * 2);
  bf16*  Wlo    = (bf16*)carve((size_t)N4 * K3 * 2);
  float* Z      = (float*)carve((size_t)Gp * N4 * 4);

  s2s_zero_f32<<<256, 256, 0, stream>>>(h, (long)G * F);
  s2s_zero_f32<<<256, 256, 0, stream>>>(c, (long)G * F);
  s2s_zero_f32<<<256, 256, 0, stream>>>(qstar, (long)G * 2 * F);
  s2s_seg_start<<<(N + 255) / 256, 256, 0, stream>>>(idx, start, N, G);

  long wtot = (long)N4 * K3;
  s2s_cvt_weights<<<(int)((wtot + 255) / 256), 256, 0, stream>>>(kern, rkern, Whi, Wlo, F, K3, N4);

  dim3 ggrid(N4 / 256, Gp / 32);
  for (int it = 0; it < 4; ++it) {
    long atot = (long)Gp * K3;
    s2s_cvt_A<<<(int)((atot + 255) / 256), 256, 0, stream>>>(qstar, h, Ahi, Alo, F, K3, G, Gp);
    s2s_lstm_gemm<<<ggrid, 128, 0, stream>>>(Ahi, Alo, Whi, Wlo, Z, K3, N4);
    long gtot = (long)G * F;
    s2s_lstm_gates<<<(int)((gtot + 255) / 256), 256, 0, stream>>>(Z, bias, h, c, qstar, F, N4, G);
    s2s_attn_dot<<<(N + 7) / 8, 256, 0, stream>>>(x, idx, h, e, N, F);
    s2s_seg_stats<<<G, 128, 0, stream>>>(e, start, emax, invden);
    s2s_readout<<<G, 256, 0, stream>>>(x, e, start, emax, invden, qstar, F);
  }
}
